// GCN_35244501631080
// MI455X (gfx1250) — matrix operand order repaired and checked by reading, not compiled, not run on hardware
//
#include <hip/hip_runtime.h>

typedef __attribute__((ext_vector_type(16))) __bf16 bf16x16;
typedef __attribute__((ext_vector_type(8)))  __bf16 bf16x8;
typedef __attribute__((ext_vector_type(8)))  float  floatx8;
typedef __attribute__((ext_vector_type(4)))  unsigned int uintx4;
typedef __attribute__((ext_vector_type(8)))  int intx8;
typedef __attribute__((ext_vector_type(4)))  int intx4;

#define BM 128
#define BN 128
#define BK 32
#define LDSS 40   // LDS row stride in bf16 elements (64B data + 16B pad = 80B)

#if defined(__has_builtin)
#if __has_builtin(__builtin_amdgcn_tensor_load_to_lds) && \
    __has_builtin(__builtin_amdgcn_s_wait_tensorcnt)
#define USE_TDM 1
#endif
#endif

// Toolchain arity detection: the therock-10.0 HIP headers (6-arg builtin)
// ship <hip/amd_detail/amd_gfx1250_TDM.h>; ROCm 7.2 (5-arg) does not.
#ifdef USE_TDM
#if defined(__has_include)
#if __has_include(<hip/amd_detail/amd_gfx1250_TDM.h>)
#define TDM_6ARG 1
#endif
#endif
#endif

// ---------------------------------------------------------------------------
// fragment load: two 8-element halves from LDS -> one 16-element WMMA operand
// ---------------------------------------------------------------------------
__device__ __forceinline__ bf16x16 ld_frag(const __bf16* p0, const __bf16* p1) {
    bf16x8 lo = *(const bf16x8*)p0;
    bf16x8 hi = *(const bf16x8*)p1;
    return __builtin_shufflevector(lo, hi, 0,1,2,3,4,5,6,7,8,9,10,11,12,13,14,15);
}

#ifdef USE_TDM
__device__ __forceinline__ unsigned rl(unsigned x) {
    return (unsigned)__builtin_amdgcn_readfirstlane((int)x);
}
// TDM: load one 16-row x 32-col bf16 tile (row-major, stride_elems between
// rows) into LDS at lds_byte, with LDS padding 16B per 64B row -> 80B rows.
__device__ __forceinline__ void tdm_load_tile16(const __bf16* g, unsigned lds_byte,
                                                int stride_elems) {
    unsigned long long ga = (unsigned long long)(const void*)g;
    uintx4 g0;
    g0[0] = rl(1u);                                    // count=1 (valid user D#)
    g0[1] = rl(lds_byte);                              // lds_addr
    g0[2] = rl((unsigned)ga);                          // global_addr[31:0]
    g0[3] = rl((unsigned)(ga >> 32) | (2u << 30));     // global_addr[56:32] | type=2
    intx8 g1;
    // data_size=1(2B) | pad_enable | pad_interval=3(16 DW) | pad_amount=3(4 DW)
    g1[0] = (int)rl((1u << 16) | (1u << 20) | (3u << 22) | (3u << 25));
    g1[1] = (int)rl(32u << 16);                        // tensor_dim0 = 32
    g1[2] = (int)rl(16u << 16);                        // tensor_dim1 = 16
    g1[3] = (int)rl(32u << 16);                        // tile_dim0 = 32
    g1[4] = (int)rl(16u);                              // tile_dim1 = 16, tile_dim2 = 0
    g1[5] = (int)rl((unsigned)stride_elems);           // tensor_dim0_stride[31:0]
    g1[6] = 0;                                         // stride hi / dim1_stride
    g1[7] = 0;
    intx4 z4 = {0, 0, 0, 0};
#ifdef TDM_6ARG
    intx8 z8 = {0, 0, 0, 0, 0, 0, 0, 0};
    __builtin_amdgcn_tensor_load_to_lds(g0, g1, z4, z4, z8, 0);
#else
    __builtin_amdgcn_tensor_load_to_lds(g0, g1, z4, z4, 0);
#endif
}
#endif

// ---------------------------------------------------------------------------
// core 128x128 block-tile GEMM mainloop (bf16 x bf16 -> f32 accum)
//   A  : row-major [M][lda], tile rows start at m0
//   Bt : "transposed B" [Ncols][ldb] (K contiguous per output column), at n0
//   K % 32 == 0, 256 threads (8 waves, 4x2 wave grid, 32x64 per wave),
//   double-buffered LDS, tiles fed by TDM (fallback: reg-staged copies)
// ---------------------------------------------------------------------------
__device__ __forceinline__ void gemm_tile(const __bf16* __restrict__ A, int lda,
                                          const __bf16* __restrict__ Bt, int ldb,
                                          int K, int m0, int n0,
                                          floatx8 acc[2][4]) {
    __shared__ __align__(16) __bf16 sA[2 * BM * LDSS];
    __shared__ __align__(16) __bf16 sB[2 * BN * LDSS];

    const int tid  = threadIdx.x;
    const int lane = tid & 31;
    const int wm0  = ((tid >> 5) & 3) * 32;   // wave M offset in block tile
    const int wn0  = ((tid >> 5) >> 2) * 64;  // wave N offset in block tile
    const int hl   = lane >> 4;               // lane half (ISA K-split)
    const int l16  = lane & 15;

    const floatx8 z = {0.f,0.f,0.f,0.f,0.f,0.f,0.f,0.f};
#pragma unroll
    for (int i = 0; i < 2; ++i)
#pragma unroll
        for (int j = 0; j < 4; ++j) acc[i][j] = z;

    const int steps = K / BK;

    // ---- compute one BK slab from LDS buffer base pointers -----------------
    auto compute = [&](const __bf16* cA, const __bf16* cB) {
        bf16x16 af[2], bfr[4];
#pragma unroll
        for (int i = 0; i < 2; ++i) {
            // A 16x32 bf16: lane half selects K base 0/8, second chunk at +16
            const __bf16* p = &cA[(wm0 + i * 16 + l16) * LDSS + hl * 8];
            af[i] = ld_frag(p, p + 16);
        }
#pragma unroll
        for (int j = 0; j < 4; ++j) {
            // B 32x16 bf16: lane half selects K base 0/8, second chunk at +16
            const __bf16* p = &cB[(wn0 + j * 16 + l16) * LDSS + hl * 8];
            bfr[j] = ld_frag(p, p + 16);
        }
#pragma unroll
        for (int i = 0; i < 2; ++i)
#pragma unroll
            for (int j = 0; j < 4; ++j)
                acc[i][j] = __builtin_amdgcn_wmma_f32_16x16x32_bf16(
                    false, af[i], false, bfr[j], (short)0, acc[i][j], false, false);
    };

#ifdef USE_TDM
    // ---- TDM-fed pipeline: wave w moves rows [16w,16w+16) of A and B tiles -
    const int wv = __builtin_amdgcn_readfirstlane(tid >> 5);
    auto issue = [&](int kk, int buf) {
        const __bf16* ga = A  + (size_t)(m0 + wv * 16) * lda + kk;
        const __bf16* gb = Bt + (size_t)(n0 + wv * 16) * ldb + kk;
        unsigned la = (unsigned)(unsigned long long)(const void*)
                      &sA[buf * BM * LDSS + wv * 16 * LDSS];
        unsigned lb = (unsigned)(unsigned long long)(const void*)
                      &sB[buf * BN * LDSS + wv * 16 * LDSS];
        tdm_load_tile16(ga, la, lda);
        tdm_load_tile16(gb, lb, ldb);
    };

    issue(0, 0);
    __builtin_amdgcn_s_wait_tensorcnt((short)0);
    __syncthreads();
    for (int i = 0; i < steps; ++i) {
        if (i + 1 < steps) issue((i + 1) * BK, (i + 1) & 1);
        compute(&sA[(i & 1) * BM * LDSS], &sB[(i & 1) * BN * LDSS]);
        if (i + 1 < steps) {
            __builtin_amdgcn_s_wait_tensorcnt((short)0);
            __syncthreads();
        }
    }
#else
    // ---- fallback: reg-staged global->LDS copies, double buffered ----------
    const int ldr = tid >> 1;                 // tile row (0..127)
    const int ldk = (tid & 1) * 16;           // k offset (0 or 16)
    const __bf16* gA = A  + (size_t)(m0 + ldr) * lda + ldk;
    const __bf16* gB = Bt + (size_t)(n0 + ldr) * ldb + ldk;
    const int woff = ldr * LDSS + ldk;

    bf16x8 a0, a1, b0, b1;
    auto ldg = [&](int kk) {
        a0 = *(const bf16x8*)(gA + kk);
        a1 = *(const bf16x8*)(gA + kk + 8);
        b0 = *(const bf16x8*)(gB + kk);
        b1 = *(const bf16x8*)(gB + kk + 8);
    };
    auto sts = [&](int buf) {
        __bf16* wA = &sA[buf * BM * LDSS + woff];
        __bf16* wB = &sB[buf * BN * LDSS + woff];
        *(bf16x8*)(wA)     = a0;  *(bf16x8*)(wA + 8) = a1;
        *(bf16x8*)(wB)     = b0;  *(bf16x8*)(wB + 8) = b1;
    };

    ldg(0);
    sts(0);
    __syncthreads();
    for (int i = 0; i < steps; ++i) {
        if (i + 1 < steps) ldg((i + 1) * BK);
        compute(&sA[(i & 1) * BM * LDSS], &sB[(i & 1) * BN * LDSS]);
        if (i + 1 < steps) {
            sts((i + 1) & 1);
            __syncthreads();
        }
    }
#endif
}

// ---------------------------------------------------------------------------
// conversions
// ---------------------------------------------------------------------------
__global__ __launch_bounds__(256) void k_cvt_w(
    const float* __restrict__ Wp, const float* __restrict__ Wt,
    const float* __restrict__ Wb, const float* __restrict__ bp,
    const float* __restrict__ bt, const float* __restrict__ bb,
    __bf16* __restrict__ Wqkv, float* __restrict__ biasC) {
    const int idx = blockIdx.x * 256 + threadIdx.x;   // 1536*512 elements
    if (idx >= 1536 * 512) return;
    const int row = idx >> 9, col = idx & 511;
    const int which = row >> 9, sr = row & 511;
    const float* src = (which == 0) ? Wp : (which == 1) ? Wt : Wb;
    Wqkv[idx] = (__bf16)src[sr * 512 + col];
    if (idx < 1536)
        biasC[idx] = (idx < 512) ? bp[idx] : (idx < 1024) ? bt[idx - 512] : bb[idx - 1024];
}

__global__ __launch_bounds__(256) void k_cvt_x(
    const float* __restrict__ X, __bf16* __restrict__ Xt) {
    const int idx = blockIdx.x * 256 + threadIdx.x;   // B*N*C
    const int c = idx & 511, n = (idx >> 9) & 1023, b = idx >> 19;
    Xt[idx] = (__bf16)X[((size_t)b * 512 + c) * 1024 + n];
}

// ---------------------------------------------------------------------------
// stage A: phi/theta/beta = Wqkv @ X  (+bias)
//   phi, theta stored transposed [B][N][C]; beta row-major [B][C][N]
// ---------------------------------------------------------------------------
__global__ __launch_bounds__(256) void k_gemm_qkv(
    const __bf16* __restrict__ Wqkv, const float* __restrict__ bias,
    const __bf16* __restrict__ Xt, __bf16* __restrict__ Pt,
    __bf16* __restrict__ Tt, __bf16* __restrict__ Brm) {
    const int b  = blockIdx.z;
    const int m0 = blockIdx.y * BM;   // 0..1536  (output channel, stacked)
    const int n0 = blockIdx.x * BN;   // 0..1024  (spatial)
    floatx8 acc[2][4];
    gemm_tile(Wqkv, 512, Xt + (size_t)b * 1024 * 512, 512, 512, m0, n0, acc);

    const int tid = threadIdx.x, lane = tid & 31;
    const int wm0 = ((tid >> 5) & 3) * 32, wn0 = ((tid >> 5) >> 2) * 64;
    const int hl = lane >> 4, l16 = lane & 15;
#pragma unroll
    for (int i = 0; i < 2; ++i)
#pragma unroll
        for (int j = 0; j < 4; ++j) {
            const int mb = m0 + wm0 + i * 16 + hl * 8;      // 8 consecutive M
            const int n  = n0 + wn0 + j * 16 + l16;
            if (m0 < 512) {            // phi -> Pt[b][n][c]
                bf16x8 v;
#pragma unroll
                for (int r = 0; r < 8; ++r) v[r] = (__bf16)(acc[i][j][r] + bias[mb + r]);
                *(bf16x8*)(Pt + ((size_t)b * 1024 + n) * 512 + mb) = v;
            } else if (m0 < 1024) {    // theta -> Tt[b][n][c]
                bf16x8 v;
#pragma unroll
                for (int r = 0; r < 8; ++r) v[r] = (__bf16)(acc[i][j][r] + bias[mb + r]);
                *(bf16x8*)(Tt + ((size_t)b * 1024 + n) * 512 + (mb - 512)) = v;
            } else {                   // beta -> Brm[b][c][n]
#pragma unroll
                for (int r = 0; r < 8; ++r)
                    Brm[((size_t)b * 512 + (mb - 1024 + r)) * 1024 + n] =
                        (__bf16)(acc[i][j][r] + bias[mb + r]);
            }
        }
}

// ---------------------------------------------------------------------------
// stage B: S[n][m] = sum_c phi[c][n] * theta[c][m]   (f32 scores)
// ---------------------------------------------------------------------------
__global__ __launch_bounds__(256) void k_gemm_scores(
    const __bf16* __restrict__ Pt, const __bf16* __restrict__ Tt,
    float* __restrict__ S) {
    const int b  = blockIdx.z;
    const int n0 = blockIdx.y * BM;   // score row (n)
    const int m0 = blockIdx.x * BN;   // score col (m)
    floatx8 acc[2][4];
    gemm_tile(Pt + (size_t)b * 1024 * 512, 512,
              Tt + (size_t)b * 1024 * 512, 512, 512, n0, m0, acc);

    const int tid = threadIdx.x, lane = tid & 31;
    const int wm0 = ((tid >> 5) & 3) * 32, wn0 = ((tid >> 5) >> 2) * 64;
    const int hl = lane >> 4, l16 = lane & 15;
    float* Sb = S + (size_t)b * 1024 * 1024;
#pragma unroll
    for (int i = 0; i < 2; ++i)
#pragma unroll
        for (int j = 0; j < 4; ++j) {
            const int nb = n0 + wm0 + i * 16 + hl * 8;
            const int m  = m0 + wn0 + j * 16 + l16;
#pragma unroll
            for (int r = 0; r < 8; ++r)
                Sb[(size_t)(nb + r) * 1024 + m] = acc[i][j][r];
        }
}

// ---------------------------------------------------------------------------
// row softmax over m; write A transposed (bf16 At[b][m][n])
// ---------------------------------------------------------------------------
__global__ __launch_bounds__(256) void k_softmax(
    const float* __restrict__ S, __bf16* __restrict__ At) {
    const int row = blockIdx.x;         // b*1024 + n
    const int b = row >> 10, n = row & 1023;
    const float* src = S + (size_t)row * 1024;
    const int tid = threadIdx.x, lane = tid & 31, wv = tid >> 5;
    float4 v = *(const float4*)(src + tid * 4);

    float mx = fmaxf(fmaxf(v.x, v.y), fmaxf(v.z, v.w));
    for (int off = 16; off > 0; off >>= 1) mx = fmaxf(mx, __shfl_xor(mx, off, 32));
    __shared__ float redm[8], reds[8];
    if (lane == 0) redm[wv] = mx;
    __syncthreads();
    mx = redm[0];
#pragma unroll
    for (int i = 1; i < 8; ++i) mx = fmaxf(mx, redm[i]);

    float e0 = __expf(v.x - mx), e1 = __expf(v.y - mx);
    float e2 = __expf(v.z - mx), e3 = __expf(v.w - mx);
    float s = e0 + e1 + e2 + e3;
    for (int off = 16; off > 0; off >>= 1) s += __shfl_xor(s, off, 32);
    if (lane == 0) reds[wv] = s;
    __syncthreads();
    s = 0.f;
#pragma unroll
    for (int i = 0; i < 8; ++i) s += reds[i];
    const float inv = 1.0f / s;

    __bf16* dst = At + (size_t)b * 1024 * 1024 + n;   // column n, stride 1024 per m
    const int m = tid * 4;
    dst[(size_t)(m + 0) * 1024] = (__bf16)(e0 * inv);
    dst[(size_t)(m + 1) * 1024] = (__bf16)(e1 * inv);
    dst[(size_t)(m + 2) * 1024] = (__bf16)(e2 * inv);
    dst[(size_t)(m + 3) * 1024] = (__bf16)(e3 * inv);
}

// ---------------------------------------------------------------------------
// stage C: Out[c][m] = sum_n beta[c][n] * A[n][m]  + X[c][m]
// ---------------------------------------------------------------------------
__global__ __launch_bounds__(256) void k_gemm_out(
    const __bf16* __restrict__ Brm, const __bf16* __restrict__ At,
    const float* __restrict__ X, float* __restrict__ Out) {
    const int b  = blockIdx.z;
    const int c0 = blockIdx.y * BM;   // channel
    const int m0 = blockIdx.x * BN;   // spatial
    floatx8 acc[2][4];
    gemm_tile(Brm + (size_t)b * 512 * 1024, 1024,
              At  + (size_t)b * 1024 * 1024, 1024, 1024, c0, m0, acc);

    const int tid = threadIdx.x, lane = tid & 31;
    const int wm0 = ((tid >> 5) & 3) * 32, wn0 = ((tid >> 5) >> 2) * 64;
    const int hl = lane >> 4, l16 = lane & 15;
#pragma unroll
    for (int i = 0; i < 2; ++i)
#pragma unroll
        for (int j = 0; j < 4; ++j) {
            const int cb  = c0 + wm0 + i * 16 + hl * 8;
            const int col = m0 + wn0 + j * 16 + l16;
#pragma unroll
            for (int r = 0; r < 8; ++r) {
                const size_t idx = ((size_t)b * 512 + cb + r) * 1024 + col;
                Out[idx] = acc[i][j][r] + X[idx];
            }
        }
}

// ---------------------------------------------------------------------------
extern "C" void kernel_launch(void* const* d_in, const int* in_sizes, int n_in,
                              void* d_out, int out_size, void* d_ws, size_t ws_size,
                              hipStream_t stream) {
    const float* X   = (const float*)d_in[0];
    const float* Wp  = (const float*)d_in[1];
    const float* bp  = (const float*)d_in[2];
    const float* Wt  = (const float*)d_in[3];
    const float* bt  = (const float*)d_in[4];
    const float* Wb  = (const float*)d_in[5];
    const float* bb  = (const float*)d_in[6];
    float* Out = (float*)d_out;
    (void)in_sizes; (void)n_in; (void)out_size; (void)ws_size;

    char* ws = (char*)d_ws;
    size_t off = 0;
    auto carve = [&](size_t bytes) {
        void* p = ws + off;
        off += (bytes + 255) & ~(size_t)255;
        return p;
    };
    __bf16* Wqkv  = (__bf16*)carve((size_t)1536 * 512 * 2);
    float*  biasC = (float*) carve((size_t)1536 * 4);
    __bf16* Xt    = (__bf16*)carve((size_t)8 * 1024 * 512 * 2);
    __bf16* Pt    = (__bf16*)carve((size_t)8 * 1024 * 512 * 2);
    __bf16* Tt    = (__bf16*)carve((size_t)8 * 1024 * 512 * 2);
    __bf16* Brm   = (__bf16*)carve((size_t)8 * 512 * 1024 * 2);
    float*  S     = (float*) carve((size_t)8 * 1024 * 1024 * 4);
    __bf16* At    = (__bf16*)carve((size_t)8 * 1024 * 1024 * 2);

    k_cvt_w<<<(1536 * 512 + 255) / 256, 256, 0, stream>>>(Wp, Wt, Wb, bp, bt, bb, Wqkv, biasC);
    k_cvt_x<<<(8 * 1024 * 512 + 255) / 256, 256, 0, stream>>>(X, Xt);

    dim3 gq(8, 12, 8);   // N/128, 1536/128, B
    k_gemm_qkv<<<gq, 256, 0, stream>>>(Wqkv, biasC, Xt, Pt, Tt, Brm);

    dim3 gs(8, 8, 8);    // m tiles, n tiles, B
    k_gemm_scores<<<gs, 256, 0, stream>>>(Pt, Tt, S);

    k_softmax<<<8 * 1024, 256, 0, stream>>>(S, At);

    dim3 go(8, 4, 8);    // m tiles, C/128, B
    k_gemm_out<<<go, 256, 0, stream>>>(Brm, At, X, Out);
}
